// Network_old_56736517980955
// MI455X (gfx1250) — compile-verified, matrix operand order checked by reading, not run on hardware
//
#include <hip/hip_runtime.h>

// ---------------------------------------------------------------------------
// SNN (LIF) forward scan on MI455X / gfx1250.
//   per step: xbf = bf16(x_t)  (tiny cvt kernel)
//             inp_h = xbf @ w_ih + s_h @ w_hh   -> LIF -> s_h
//             inp_o = s_h @ w_ho                -> LIF -> y[t]
// bf16 WMMA (v_wmma_f32_16x16x32_bf16), fp32 accum, LIF fused in epilogue.
// Hidden GEMM: 64x64 block tile, 4 waves, each wave 32x32 (4 acc tiles).
// Staging uses named scalar uint4 temporaries (promotable -> no scratch).
// Output GEMM: split-K across 8 waves + LDS reduction.
// ---------------------------------------------------------------------------

typedef __attribute__((ext_vector_type(16))) __bf16 v16bf;
typedef __attribute__((ext_vector_type(8)))  float  v8f;
typedef __attribute__((ext_vector_type(2)))  float  vf2;
typedef __attribute__((ext_vector_type(2)))  __bf16 vbf2;

union FragU {
  v16bf v;
  unsigned int u[8];
};

union CvtU {
  vbf2 h;
  unsigned int u;
};

// Problem constants (mirror reference)
#define NT     100
#define NB     256
#define NIN    784
#define KIN_P  800      // NIN padded to multiple of 32
#define NH     2048
#define NOUT   10
#define NOUT_P 16       // padded to one WMMA tile

#define LDS_PITCH 40    // bf16 per LDS row: 80B = 5x16B -> b128-aligned rows

// workspace byte offsets (all 256B aligned)
#define OFF_WIHT 0u                    // [NH][KIN_P] bf16  : 3,276,800
#define OFF_WHHT 3276800u              // [NH][NH]    bf16  : 8,388,608
#define OFF_WHOT 11665408u             // [NOUT_P][NH] bf16 : 65,536
#define OFF_S0   11730944u             // [NB][NH] bf16     : 1,048,576
#define OFF_S1   12779520u             // [NB][NH] bf16     : 1,048,576
#define OFF_VH   13828096u             // [NB][NH] f32      : 2,097,152
#define OFF_RH   15925248u             // [NB][NH] f32      : 2,097,152
#define OFF_VO   18022400u             // [NB][NOUT_P] f32  : 16,384
#define OFF_RO   18038784u             // [NB][NOUT_P] f32  : 16,384
#define OFF_XBF  18055168u             // [NB][KIN_P] bf16  : 409,600

__device__ __forceinline__ unsigned short f2bf(float x) {
  unsigned int u = __float_as_uint(x);
  u = (u + 0x7FFFu + ((u >> 16) & 1u)) >> 16;   // round-to-nearest-even
  return (unsigned short)u;
}

__device__ __forceinline__ unsigned int pack2bf(float a, float b) {
  CvtU c;
  vf2 f; f.x = a; f.y = b;
  c.h = __builtin_convertvector(f, vbf2);       // v_cvt_pk_bf16_f32 on gfx1250
  return c.u;
}

// ---------------------------------------------------------------------------
// Prep: bf16-convert + transpose weights, zero state. Runs once per launch.
// ---------------------------------------------------------------------------
__global__ __launch_bounds__(256) void snn_prep(
    const float* __restrict__ w_ih,   // [NIN][NH]
    const float* __restrict__ w_hh,   // [NH][NH]
    const float* __restrict__ w_ho,   // [NH][NOUT]
    unsigned short* __restrict__ wihT,  // [NH][KIN_P]
    unsigned short* __restrict__ whhT,  // [NH][NH]
    unsigned short* __restrict__ whoT,  // [NOUT_P][NH]
    float* __restrict__ v_h, float* __restrict__ r_h,
    unsigned short* __restrict__ s0,
    float* __restrict__ v_o, float* __restrict__ r_o)
{
  const unsigned int idx = blockIdx.x * 256u + threadIdx.x;

  if (idx < (unsigned)NH * NH) {        // whhT[n][k] = bf16(w_hh[k][n])
    const int n = idx / NH, k = idx % NH;
    whhT[idx] = f2bf(w_hh[(size_t)k * NH + n]);
  }
  if (idx < (unsigned)NH * KIN_P) {     // wihT[n][k] (zero pad k >= NIN)
    const int n = idx / KIN_P, k = idx % KIN_P;
    wihT[idx] = (k < NIN) ? f2bf(w_ih[(size_t)k * NH + n]) : (unsigned short)0;
  }
  if (idx < (unsigned)NOUT_P * NH) {    // whoT[n][k] (zero pad n >= NOUT)
    const int n = idx / NH, k = idx % NH;
    whoT[idx] = (n < NOUT) ? f2bf(w_ho[(size_t)k * NOUT + n]) : (unsigned short)0;
  }
  if (idx < (unsigned)NB * NH) {        // zero hidden state
    v_h[idx] = 0.0f; r_h[idx] = 0.0f; s0[idx] = 0;
  }
  if (idx < (unsigned)NB * NOUT_P) {    // zero output state
    v_o[idx] = 0.0f; r_o[idx] = 0.0f;
  }
}

// ---------------------------------------------------------------------------
// Per-step input conversion: xbf[b][k] = bf16(x_t[b][k]), zero pad k>=NIN.
// ---------------------------------------------------------------------------
__global__ __launch_bounds__(256) void snn_xcvt(
    const float* __restrict__ x_t,            // [NB][NIN]
    unsigned short* __restrict__ xbf)         // [NB][KIN_P]
{
  const int gid = blockIdx.x * 256 + threadIdx.x;   // 0..25599
#pragma unroll
  for (int i = 0; i < 2; ++i) {
    const int q  = gid + 25600 * i;           // float4 index over [NB][200]
    const int b  = q / 200;
    const int kq = (q % 200) << 2;            // 0,4,...,796
    float4 xv;
    if (kq < NIN) xv = *(const float4*)&x_t[(size_t)b * NIN + kq];
    else          xv = make_float4(0.f, 0.f, 0.f, 0.f);
    uint2 p;
    p.x = pack2bf(xv.x, xv.y);
    p.y = pack2bf(xv.z, xv.w);
    *(uint2*)&xbf[(size_t)b * KIN_P + kq] = p;
  }
}

// ---------------------------------------------------------------------------
// Hidden layer: inp = xbf @ w_ih + s_old @ w_hh, fused LIF -> s_new, v_h, r_h
// grid (NH/64, NB/64) = (32, 4) = 128 blocks, 128 threads (4 waves).
// Each wave: 32x32 output = 4 acc tiles; 8 ds_load_b128 per 4 WMMA.
// ---------------------------------------------------------------------------
__global__ __launch_bounds__(128, 1) void snn_hidden(
    const unsigned short* __restrict__ xbf,   // [NB][KIN_P] bf16, this step
    const unsigned short* __restrict__ wihT,  // [NH][KIN_P] bf16
    const unsigned short* __restrict__ whhT,  // [NH][NH]    bf16
    const unsigned short* __restrict__ s_old, // [NB][NH]    bf16
    unsigned short* __restrict__ s_new,       // [NB][NH]    bf16
    float* __restrict__ v_h, float* __restrict__ r_h)
{
  __shared__ __align__(16) unsigned short ldsA[2][64 * LDS_PITCH];
  __shared__ __align__(16) unsigned short ldsB[2][64 * LDS_PITCH];

  const int tid  = threadIdx.x;
  const int lane = tid & 31;
  const int wave = tid >> 5;       // 0..3
  const int half = lane >> 4;
  const int row  = lane & 15;
  const int wm   = wave & 1;       // 2 m-groups of 32
  const int wn   = wave >> 1;      // 2 n-groups of 32
  const int m0   = blockIdx.y * 64;
  const int n0   = blockIdx.x * 64;

  v8f acc[2][2];                   // [s = m-subtile][t = n-subtile]
#pragma unroll
  for (int s = 0; s < 2; ++s)
#pragma unroll
    for (int t = 0; t < 2; ++t)
#pragma unroll
      for (int e = 0; e < 8; ++e) acc[s][t][e] = 0.0f;

  auto compute = [&](int buf) {
    FragU a[2];
#pragma unroll
    for (int s = 0; s < 2; ++s) {
      const unsigned short* pa =
          &ldsA[buf][(wm * 32 + s * 16 + row) * LDS_PITCH + (half << 3)];
      *(uint4*)&a[s].u[0] = *(const uint4*)pa;
      *(uint4*)&a[s].u[4] = *(const uint4*)(pa + 16);
    }
#pragma unroll
    for (int t = 0; t < 2; ++t) {
      FragU b;
      const unsigned short* pb =
          &ldsB[buf][(wn * 32 + t * 16 + row) * LDS_PITCH + (half << 3)];
      *(uint4*)&b.u[0] = *(const uint4*)pb;
      *(uint4*)&b.u[4] = *(const uint4*)(pb + 16);
#pragma unroll
      for (int s = 0; s < 2; ++s)
        acc[s][t] = __builtin_amdgcn_wmma_f32_16x16x32_bf16(
            false, a[s].v, false, b.v, (short)0, acc[s][t], false, false);
    }
  };

  const int c_ri = tid >> 2;            // staging row (0..31 base)
  const int c_ko = (tid & 3) << 3;      // staging k offset 0,8,16,24
  const int st0  = c_ri * LDS_PITCH + c_ko;          // LDS staging slots
  const int st1  = (c_ri + 32) * LDS_PITCH + c_ko;

  // Both phases: identical bf16 b128 double-buffered pipeline.
  // Named scalar temporaries only -> SROA keeps staging in registers.
#define PHASE(ASRC, APITCH, BSRC, BPITCH, KTOT)                               \
  {                                                                           \
    const unsigned short* Ab = (ASRC) + (size_t)m0 * (APITCH);                \
    const unsigned short* Bb = (BSRC) + (size_t)n0 * (BPITCH);                \
    const size_t ga0 = (size_t)c_ri * (APITCH) + c_ko;                        \
    const size_t ga1 = (size_t)(c_ri + 32) * (APITCH) + c_ko;                 \
    const size_t gb0 = (size_t)c_ri * (BPITCH) + c_ko;                        \
    const size_t gb1 = (size_t)(c_ri + 32) * (BPITCH) + c_ko;                 \
    uint4 a0 = *(const uint4*)(Ab + ga0);                                     \
    uint4 a1 = *(const uint4*)(Ab + ga1);                                     \
    uint4 b0 = *(const uint4*)(Bb + gb0);                                     \
    uint4 b1 = *(const uint4*)(Bb + gb1);                                     \
    __syncthreads();                                                          \
    *(uint4*)&ldsA[0][st0] = a0;                                              \
    *(uint4*)&ldsA[0][st1] = a1;                                              \
    *(uint4*)&ldsB[0][st0] = b0;                                              \
    *(uint4*)&ldsB[0][st1] = b1;                                              \
    int buf = 0;                                                              \
    for (int k0 = 0; k0 < (KTOT); k0 += 32) {                                 \
      __syncthreads();                                                        \
      const bool more = (k0 + 32 < (KTOT));                                   \
      if (more) {                                                             \
        a0 = *(const uint4*)(Ab + ga0 + k0 + 32);                             \
        a1 = *(const uint4*)(Ab + ga1 + k0 + 32);                             \
        b0 = *(const uint4*)(Bb + gb0 + k0 + 32);                             \
        b1 = *(const uint4*)(Bb + gb1 + k0 + 32);                             \
        __builtin_prefetch(Bb + gb0 + k0 + 64, 0, 3);                         \
      }                                                                       \
      compute(buf);                                                           \
      if (more) {                                                             \
        const int nb = buf ^ 1;                                               \
        *(uint4*)&ldsA[nb][st0] = a0;                                         \
        *(uint4*)&ldsA[nb][st1] = a1;                                         \
        *(uint4*)&ldsB[nb][st0] = b0;                                         \
        *(uint4*)&ldsB[nb][st1] = b1;                                         \
      }                                                                       \
      buf ^= 1;                                                               \
    }                                                                         \
  }

  PHASE(xbf,   KIN_P, wihT, KIN_P, KIN_P)   // phase 1: x @ w_ih
  PHASE(s_old, NH,    whhT, NH,    NH)      // phase 2: s @ w_hh

#undef PHASE

  // ---------------- fused LIF epilogue -------------------------------------
  // C/D tile layout: lanes 0-15: n=lane, m=e; lanes 16-31: n=lane-16, m=8+e
#pragma unroll
  for (int s = 0; s < 2; ++s) {
#pragma unroll
    for (int t = 0; t < 2; ++t) {
#pragma unroll
      for (int e = 0; e < 8; ++e) {
        const int gb = m0 + wm * 32 + s * 16 + half * 8 + e;
        const int gj = n0 + wn * 32 + t * 16 + row;
        const size_t idx = (size_t)gb * NH + gj;
        const float inp = acc[s][t][e];
        float vv = v_h[idx];
        float rr = r_h[idx];
        vv *= 0.9f;                              // v - DT*decay*(v - rest)
        rr = fmaxf(rr - 1.0f, 0.0f);             // refractory tick (DT=1)
        const bool sp = (vv >= 1.0f) && (rr == 0.0f);
        if (sp) { rr = 2.0f; vv = 0.0f; }        // spike: reset + refractory
        if (rr == 0.0f) vv += inp;               // integrate if not refractory
        v_h[idx] = vv;
        r_h[idx] = rr;
        s_new[idx] = sp ? (unsigned short)0x3F80 : (unsigned short)0;
      }
    }
  }
}

// ---------------------------------------------------------------------------
// Output layer: inp_o = s_new @ w_ho (N padded 10->16), fused LIF, y[t].
// grid 16 blocks (one 16-row M-tile each), 8 waves split-K (256-wide chunks),
// partials reduced through LDS, then LIF.
// ---------------------------------------------------------------------------
__global__ __launch_bounds__(256, 1) void snn_out(
    const unsigned short* __restrict__ s_new, // [NB][NH] bf16
    const unsigned short* __restrict__ whoT,  // [NOUT_P][NH] bf16
    float* __restrict__ v_o, float* __restrict__ r_o, // [NB][NOUT_P]
    float* __restrict__ y_t)                  // [NB][NOUT] fp32
{
  __shared__ __align__(16) float red[8 * 256];

  const int tid  = threadIdx.x;
  const int lane = tid & 31;
  const int wave = tid >> 5;        // 0..7 -> K chunk
  const int half = lane >> 4;
  const int row  = lane & 15;
  const int m0   = blockIdx.x * 16;

  v8f acc;
#pragma unroll
  for (int e = 0; e < 8; ++e) acc[e] = 0.0f;

  const size_t brow = (size_t)row * NH;
  const size_t arow = (size_t)(m0 + row) * NH;
  const int kbeg = wave * (NH / 8);

  for (int k0 = kbeg; k0 < kbeg + NH / 8; k0 += 32) {
    const int kb = k0 + (half << 3);
    FragU a, b;
    *(uint4*)&b.u[0] = *(const uint4*)&whoT[brow + kb];
    *(uint4*)&b.u[4] = *(const uint4*)&whoT[brow + kb + 16];
    *(uint4*)&a.u[0] = *(const uint4*)&s_new[arow + kb];
    *(uint4*)&a.u[4] = *(const uint4*)&s_new[arow + kb + 16];
    acc = __builtin_amdgcn_wmma_f32_16x16x32_bf16(
        false, a.v, false, b.v, (short)0, acc, false, false);
  }

  // dump per-wave partial tile: element (m = half*8+e, n = row)
#pragma unroll
  for (int e = 0; e < 8; ++e)
    red[wave * 256 + (half * 8 + e) * 16 + row] = acc[e];
  __syncthreads();

  // one thread per output element: reduce 8 partials, then LIF
  const int n = tid & 15;
  float inp = 0.0f;
#pragma unroll
  for (int w = 0; w < 8; ++w) inp += red[w * 256 + tid];

  const int gb = m0 + (tid >> 4);
  const int idx = gb * NOUT_P + n;
  float vv = v_o[idx];
  float rr = r_o[idx];
  vv *= 0.9f;
  rr = fmaxf(rr - 1.0f, 0.0f);
  const bool sp = (vv >= 1.0f) && (rr == 0.0f);
  if (sp) { rr = 2.0f; vv = 0.0f; }
  if (rr == 0.0f) vv += inp;
  v_o[idx] = vv;
  r_o[idx] = rr;
  if (n < NOUT) y_t[gb * NOUT + n] = sp ? 1.0f : 0.0f;
}

// ---------------------------------------------------------------------------
extern "C" void kernel_launch(void* const* d_in, const int* in_sizes, int n_in,
                              void* d_out, int out_size, void* d_ws, size_t ws_size,
                              hipStream_t stream) {
  (void)in_sizes; (void)n_in; (void)out_size; (void)ws_size;

  const float* x_in = (const float*)d_in[0];   // [NT][NB][NIN]
  const float* w_ih = (const float*)d_in[1];   // [NIN][NH]
  const float* w_hh = (const float*)d_in[2];   // [NH][NH]
  const float* w_ho = (const float*)d_in[3];   // [NH][NOUT]
  float* out = (float*)d_out;                  // [NT][NB][NOUT]

  char* ws = (char*)d_ws;
  unsigned short* wihT = (unsigned short*)(ws + OFF_WIHT);
  unsigned short* whhT = (unsigned short*)(ws + OFF_WHHT);
  unsigned short* whoT = (unsigned short*)(ws + OFF_WHOT);
  unsigned short* s0   = (unsigned short*)(ws + OFF_S0);
  unsigned short* s1   = (unsigned short*)(ws + OFF_S1);
  float* v_h = (float*)(ws + OFF_VH);
  float* r_h = (float*)(ws + OFF_RH);
  float* v_o = (float*)(ws + OFF_VO);
  float* r_o = (float*)(ws + OFF_RO);
  unsigned short* xbf = (unsigned short*)(ws + OFF_XBF);

  snn_prep<<<16384, 256, 0, stream>>>(w_ih, w_hh, w_ho, wihT, whhT, whoT,
                                      v_h, r_h, s0, v_o, r_o);

  for (int t = 0; t < NT; ++t) {
    const unsigned short* s_old = (t & 1) ? s1 : s0;
    unsigned short*       s_new = (t & 1) ? s0 : s1;
    snn_xcvt<<<100, 256, 0, stream>>>(x_in + (size_t)t * NB * NIN, xbf);
    snn_hidden<<<dim3(NH / 64, NB / 64), 128, 0, stream>>>(
        xbf, wihT, whhT, s_old, s_new, v_h, r_h);
    snn_out<<<16, 256, 0, stream>>>(s_new, whoT, v_o, r_o,
                                    out + (size_t)t * NB * NOUT);
  }
}
